// Attention_29480655519761
// MI455X (gfx1250) — matrix operand order repaired and checked by reading, not compiled, not run on hardware
//
#include <hip/hip_runtime.h>
#include <hip/hip_bf16.h>

// MI455X (gfx1250) wave32 + WMMA bf16 implementation of MHA forward.
// B=2, N=2048, C=768, H=12, HD=64.
// Workspace layout (bf16): q | k | v | attn  (each B*H*N*HD = 3,145,728 elems)
// CDNA5 paths: v_wmma_f32_16x16x32_bf16 for all matmuls; TDM tensor_load_to_lds
// (TENSORcnt) for flash K/V tiles with hardware LDS padding; async-to-LDS
// (ASYNCcnt) for proj A tiles; global_prefetch for the fp32-source GEMM.

#define Bsz 2
#define Nsq 2048
#define Cch 768
#define Hh  12
#define HD  64

typedef __bf16 bf16;
typedef __attribute__((ext_vector_type(16))) __bf16 bf16x16;
typedef __attribute__((ext_vector_type(8)))  float  f32x8;
typedef unsigned int u32x4 __attribute__((ext_vector_type(4)));
typedef unsigned int u32x8 __attribute__((ext_vector_type(8)));

typedef __attribute__((address_space(3))) void lds_void_t;

__device__ __forceinline__ unsigned int lds_off(const void* p) {
  // generic -> LDS addrspace -> 32-bit LDS byte offset
  return (unsigned int)(uintptr_t)(lds_void_t*)p;
}

// Async 16-byte copy global -> LDS (per active lane). Tracked by ASYNCcnt.
__device__ __forceinline__ void async_copy16(void* lds_dst, const void* gsrc) {
  asm volatile("global_load_async_to_lds_b128 %0, %1, off"
               :
               : "v"(lds_off(lds_dst)), "v"((unsigned long long)(uintptr_t)gsrc)
               : "memory");
}
__device__ __forceinline__ void wait_async0() {
  asm volatile("s_wait_asynccnt 0x0" ::: "memory");
}

// TDM: DMA one 32x64 bf16 tile from a [2048][64] bf16 tensor into LDS, with
// hardware row padding of 16B per 128B row -> matches a [32][72] bf16 layout.
// 2-group descriptor (<=2D tensor): VADDR2/VADDR3 omitted (NULL).
__device__ __forceinline__ void tdm_load_tile_32x64(unsigned int lds_byte,
                                                    const bf16* gsrc) {
  const unsigned long long ga = (unsigned long long)(uintptr_t)gsrc;
  u32x4 g0;
  g0[0] = 1u;                                     // count=1 (valid user D#)
  g0[1] = lds_byte;                               // lds_addr
  g0[2] = (unsigned int)ga;                       // global_addr[31:0]
  g0[3] = (unsigned int)(ga >> 32) | (2u << 30);  // global_addr[56:32] | type=2
  u32x8 g1;
  g1[0] = 0x07110000u;  // data_size=2B; pad_enable; pad_interval=32dw; pad=4dw
  g1[1] = 64u << 16;    // tensor_dim0 = 64 elems     (bits  79:48)
  g1[2] = 2048u << 16;  // tensor_dim1 = 2048 rows    (bits 111:80)
  g1[3] = 64u << 16;    // tile_dim0   = 64           (bits 127:112)
  g1[4] = 32u;          // tile_dim1   = 32           (bits 143:128)
  g1[5] = 64u;          // tensor_dim0_stride = 64    (bits 207:160)
  g1[6] = 0u;
  g1[7] = 0u;
  asm volatile("tensor_load_to_lds %0, %1" :: "s"(g0), "s"(g1) : "memory");
}

__device__ __forceinline__ f32x8 wmma_bf16(bf16x16 a, bf16x16 b, f32x8 c) {
  // D = A(16x32 bf16) * B(32x16 bf16) + C(16x16 f32)
  return __builtin_amdgcn_wmma_f32_16x16x32_bf16(false, a, false, b, (short)0, c,
                                                 false, false);
}

// A-fragment (16x32 bf16, MxK) from row-major [16][stride] source.
__device__ __forceinline__ bf16x16 load_a_frag(const bf16* base, int stride) {
  const int lane = threadIdx.x & 31;
  const int m    = lane & 15;
  const int koff = (lane >> 4) * 16;
  const bf16* row = base + m * stride;
  bf16x16 a;
#pragma unroll
  for (int i = 0; i < 8; ++i) {
    a[i]     = row[koff + i];
    a[8 + i] = row[koff + 8 + i];
  }
  return a;
}

// B-fragment (32x16 bf16, KxN) from an [N][K] row-major source (B(k,n)=src[n][k]).
__device__ __forceinline__ bf16x16 load_b_frag_nk(const bf16* base, int stride) {
  const int lane = threadIdx.x & 31;
  const int n    = lane & 15;
  const int koff = (lane >> 4) * 16;
  const bf16* row = base + n * stride;
  bf16x16 b;
#pragma unroll
  for (int v = 0; v < 8; ++v) {
    b[2 * v]     = row[koff + 2 * v];
    b[2 * v + 1] = row[koff + 2 * v + 1];
  }
  return b;
}

// B-fragment (32x16 bf16, KxN) from a [K][N] row-major source (B(k,n)=src[k][n]).
__device__ __forceinline__ bf16x16 load_b_frag_kn(const bf16* base, int stride) {
  const int lane = threadIdx.x & 31;
  const int n    = lane & 15;
  const int koff = (lane >> 4) * 16;
  bf16x16 b;
#pragma unroll
  for (int v = 0; v < 8; ++v) {
    b[2 * v]     = base[(koff + 2 * v) * stride + n];
    b[2 * v + 1] = base[(koff + 2 * v + 1) * stride + n];
  }
  return b;
}

// ---------------------------------------------------------------------------
// Kernel 1: qkv = x @ W_qkv^T + b_qkv, scattered to q/k/v [B,H,N,HD] bf16.
// M = B*N = 4096, Ncols = 3C = 2304, K = 768. Scale*log2e folded into q.
// ---------------------------------------------------------------------------
__global__ __launch_bounds__(256) void qkv_gemm_kernel(
    const float* __restrict__ X, const float* __restrict__ Wq,
    const float* __restrict__ bq,
    bf16* __restrict__ qo, bf16* __restrict__ ko, bf16* __restrict__ vo) {
  __shared__ bf16 As[128][40];  // [m][k], padded
  __shared__ bf16 Bs[128][40];  // [n][k], padded

  const int m0   = blockIdx.y * 128;
  const int n0   = blockIdx.x * 128;
  const int w    = threadIdx.x >> 5;
  const int lane = threadIdx.x & 31;
  const int wm   = (w & 1) * 64;   // 2 waves along M (64 rows each)
  const int wn   = (w >> 1) * 32;  // 4 waves along N (32 cols each)

  f32x8 acc[4][2];
#pragma unroll
  for (int mi = 0; mi < 4; ++mi)
#pragma unroll
    for (int ni = 0; ni < 2; ++ni) acc[mi][ni] = {};

  const int r0 = threadIdx.x >> 5;  // 0..7
  const int c0 = threadIdx.x & 31;  // 0..31

  for (int kt = 0; kt < 24; ++kt) {
    const int k0 = kt * 32;
    __syncthreads();
#pragma unroll
    for (int i = 0; i < 16; ++i) {
      const int r = r0 + i * 8;
      As[r][c0] = (bf16)X[(size_t)(m0 + r) * Cch + k0 + c0];
      Bs[r][c0] = (bf16)Wq[(size_t)(n0 + r) * Cch + k0 + c0];
    }
    if (kt < 23) {  // hint next k-tile into cache (global_prefetch_b8)
      __builtin_prefetch(&X[(size_t)(m0 + r0) * Cch + k0 + 32 + c0], 0, 0);
      __builtin_prefetch(&Wq[(size_t)(n0 + r0) * Cch + k0 + 32 + c0], 0, 0);
    }
    __syncthreads();

    bf16x16 af[4], bfr[2];
#pragma unroll
    for (int mi = 0; mi < 4; ++mi) af[mi] = load_a_frag(&As[wm + mi * 16][0], 40);
#pragma unroll
    for (int ni = 0; ni < 2; ++ni) bfr[ni] = load_b_frag_nk(&Bs[wn + ni * 16][0], 40);
#pragma unroll
    for (int mi = 0; mi < 4; ++mi)
#pragma unroll
      for (int ni = 0; ni < 2; ++ni)
        acc[mi][ni] = wmma_bf16(af[mi], bfr[ni], acc[mi][ni]);
  }

  // Epilogue: bias + scatter to q/k/v in [B,H,N,HD] bf16.
  const int   half = lane >> 4;
  const int   ncol = lane & 15;
  const float QS   = 0.125f * 1.44269504088896f;  // HD^-0.5 * log2(e)
#pragma unroll
  for (int ni = 0; ni < 2; ++ni) {
    const int ocb  = n0 + wn + ni * 16;      // 16-aligned -> selector uniform
    const int wch  = ocb / Cch;              // 0=q 1=k 2=v
    const int remb = ocb - wch * Cch;
    const int rem  = remb + ncol;
    const int h    = rem >> 6;
    const int d    = rem & 63;
    const float bias = bq[ocb + ncol];
#pragma unroll
    for (int mi = 0; mi < 4; ++mi)
#pragma unroll
      for (int r = 0; r < 8; ++r) {
        const int row = m0 + wm + mi * 16 + half * 8 + r;  // [0,4096)
        const int bb  = row >> 11;
        const int n   = row & 2047;
        const float val = acc[mi][ni][r] + bias;
        const size_t idx = ((size_t)(bb * Hh + h) * Nsq + n) * HD + d;
        if (wch == 0)      qo[idx] = (bf16)(val * QS);
        else if (wch == 1) ko[idx] = (bf16)val;
        else               vo[idx] = (bf16)val;
      }
  }
}

// ---------------------------------------------------------------------------
// Kernel 2: flash attention per (b,h). Block = 128 q-rows, wave = 16 q-rows.
// K/V tiles (32x64 bf16) double-buffered in LDS, DMA'd by the Tensor Data
// Mover (one tensor_load_to_lds per tile, issued by wave 0, TENSORcnt-waited).
// Online softmax in exp2 domain (scale*log2e already folded into q).
// ---------------------------------------------------------------------------
__global__ __launch_bounds__(256) void flash_attn_kernel(
    const bf16* __restrict__ q, const bf16* __restrict__ k,
    const bf16* __restrict__ v, bf16* __restrict__ attn) {
  __shared__ bf16 Klds[2][32][72];     // [buf][kv][d] -- TDM pads 16B per row
  __shared__ bf16 Vlds[2][32][72];
  __shared__ bf16 Plds[8][16][40];     // per-wave 16 x 32 P scratch (padded)

  const int w    = threadIdx.x >> 5;
  const int lane = threadIdx.x & 31;
  const int bh   = blockIdx.y;                  // b*H + h
  const int q0   = blockIdx.x * 128 + w * 16;   // this wave's q-row base
  const int ncol = lane & 15;

  const bf16* Kbh = k + (size_t)bh * Nsq * HD;
  const bf16* Vbh = v + (size_t)bh * Nsq * HD;

  // issue tile 0 into buffer 0 (wave 0's TENSORcnt = 2)
  if (w == 0) {
    tdm_load_tile_32x64(lds_off(&Klds[0][0][0]), Kbh);
    tdm_load_tile_32x64(lds_off(&Vlds[0][0][0]), Vbh);
  }

  // Q strip (16 x 64) held in registers as two A-fragments (d 0..31, 32..63).
  const bf16* qbase = q + ((size_t)bh * Nsq + q0) * HD;
  bf16x16 qf0 = load_a_frag(qbase, HD);
  bf16x16 qf1 = load_a_frag(qbase + 32, HD);

  float mrow[8], lrow[8];
  f32x8 o[4];
#pragma unroll
  for (int r = 0; r < 8; ++r) { mrow[r] = -3.0e38f; lrow[r] = 0.0f; }
#pragma unroll
  for (int t = 0; t < 4; ++t) o[t] = {};

  const int T = Nsq / 32;
  for (int kt = 0; kt < T; ++kt) {
    const int cur = kt & 1;
    if (w == 0) {
      if (kt + 1 < T) {
        // pipeline: issue tile kt+1 into the other buffer, then wait only for
        // tile kt (TENSORcnt down to 2 leaves the new pair in flight)
        const size_t nb = (size_t)(kt + 1) * 32 * HD;
        tdm_load_tile_32x64(lds_off(&Klds[cur ^ 1][0][0]), Kbh + nb);
        tdm_load_tile_32x64(lds_off(&Vlds[cur ^ 1][0][0]), Vbh + nb);
        __builtin_amdgcn_s_wait_tensorcnt((short)2);
      } else {
        __builtin_amdgcn_s_wait_tensorcnt((short)0);
      }
    }
    __syncthreads();  // publish tile kt (wave 0 arrives only after TDM done)

    // S = Q * K^T : two 16x16 tiles (kv 0..15, 16..31), reduce over d=64.
    f32x8 s0 = {}, s1 = {};
    s0 = wmma_bf16(qf0, load_b_frag_nk(&Klds[cur][0][0],  72), s0);
    s0 = wmma_bf16(qf1, load_b_frag_nk(&Klds[cur][0][32], 72), s0);
    s1 = wmma_bf16(qf0, load_b_frag_nk(&Klds[cur][16][0],  72), s1);
    s1 = wmma_bf16(qf1, load_b_frag_nk(&Klds[cur][16][32], 72), s1);

    // Online softmax: rows live across 16 lanes of each half-wave.
#pragma unroll
    for (int r = 0; r < 8; ++r) {
      float t = fmaxf(s0[r], s1[r]);
      t = fmaxf(t, __shfl_xor(t, 1, 32));
      t = fmaxf(t, __shfl_xor(t, 2, 32));
      t = fmaxf(t, __shfl_xor(t, 4, 32));
      t = fmaxf(t, __shfl_xor(t, 8, 32));
      const float mnew  = fmaxf(mrow[r], t);
      const float alpha = exp2f(mrow[r] - mnew);
      const float p0    = exp2f(s0[r] - mnew);
      const float p1    = exp2f(s1[r] - mnew);
      float rs = p0 + p1;
      rs += __shfl_xor(rs, 1, 32);
      rs += __shfl_xor(rs, 2, 32);
      rs += __shfl_xor(rs, 4, 32);
      rs += __shfl_xor(rs, 8, 32);
      lrow[r] = lrow[r] * alpha + rs;
      mrow[r] = mnew;
      o[0][r] *= alpha; o[1][r] *= alpha; o[2][r] *= alpha; o[3][r] *= alpha;
      const int m = ((lane >> 4) * 8) + r;  // row within 16
      Plds[w][m][ncol]      = (bf16)p0;
      Plds[w][m][16 + ncol] = (bf16)p1;
    }

    // O += P(16x32) * V(32x64). Same-wave DS ops are in-order -> no barrier.
    bf16x16 pf = load_a_frag(&Plds[w][0][0], 40);
    o[0] = wmma_bf16(pf, load_b_frag_kn(&Vlds[cur][0][0],  72), o[0]);
    o[1] = wmma_bf16(pf, load_b_frag_kn(&Vlds[cur][0][16], 72), o[1]);
    o[2] = wmma_bf16(pf, load_b_frag_kn(&Vlds[cur][0][32], 72), o[2]);
    o[3] = wmma_bf16(pf, load_b_frag_kn(&Vlds[cur][0][48], 72), o[3]);

    __syncthreads();  // buffer `cur` may be re-filled by next iteration's TDM
  }

  // Normalize and write to attn [B,N,C] bf16 (heads concatenated).
  const int half = lane >> 4;
  const int h    = bh % Hh;
  const int bb   = bh / Hh;
#pragma unroll
  for (int t = 0; t < 4; ++t)
#pragma unroll
    for (int r = 0; r < 8; ++r) {
      const int   qrow = q0 + half * 8 + r;
      const int   d    = t * 16 + ncol;
      const float oval = o[t][r] / lrow[r];
      attn[((size_t)(bb * Nsq + qrow)) * Cch + h * HD + d] = (bf16)oval;
    }
}

// ---------------------------------------------------------------------------
// Kernel 3: out = attn @ W_proj^T + b_proj. M=4096, Ncols=768, K=768, fp32 out.
// A-tile (bf16) staged with async-to-LDS; B-tile converted fp32->bf16 in VGPRs.
// ---------------------------------------------------------------------------
__global__ __launch_bounds__(256) void proj_gemm_kernel(
    const bf16* __restrict__ A, const float* __restrict__ Wp,
    const float* __restrict__ bp, float* __restrict__ out) {
  __shared__ bf16 As[128][40];
  __shared__ bf16 Bs[128][40];

  const int m0   = blockIdx.y * 128;
  const int n0   = blockIdx.x * 128;
  const int w    = threadIdx.x >> 5;
  const int lane = threadIdx.x & 31;
  const int wm   = (w & 1) * 64;
  const int wn   = (w >> 1) * 32;

  f32x8 acc[4][2];
#pragma unroll
  for (int mi = 0; mi < 4; ++mi)
#pragma unroll
    for (int ni = 0; ni < 2; ++ni) acc[mi][ni] = {};

  const int r0 = threadIdx.x >> 5;
  const int c0 = threadIdx.x & 31;

  for (int kt = 0; kt < 24; ++kt) {
    const int k0 = kt * 32;
    __syncthreads();
    // A tile: 128x32 bf16 = 512 x 16B chunks, 2 per thread, async to LDS
#pragma unroll
    for (int j = 0; j < 2; ++j) {
      const int ch = threadIdx.x + j * 256;
      const int r  = ch >> 2;
      const int c  = (ch & 3) * 8;
      async_copy16(&As[r][c], A + (size_t)(m0 + r) * Cch + k0 + c);
    }
    // B tile: fp32 -> bf16 conversion path
#pragma unroll
    for (int i = 0; i < 16; ++i) {
      const int r = r0 + i * 8;
      Bs[r][c0] = (bf16)Wp[(size_t)(n0 + r) * Cch + k0 + c0];
    }
    if (kt < 23)
      __builtin_prefetch(&Wp[(size_t)(n0 + r0) * Cch + k0 + 32 + c0], 0, 0);
    wait_async0();
    __syncthreads();

    bf16x16 af[4], bfr[2];
#pragma unroll
    for (int mi = 0; mi < 4; ++mi) af[mi] = load_a_frag(&As[wm + mi * 16][0], 40);
#pragma unroll
    for (int ni = 0; ni < 2; ++ni) bfr[ni] = load_b_frag_nk(&Bs[wn + ni * 16][0], 40);
#pragma unroll
    for (int mi = 0; mi < 4; ++mi)
#pragma unroll
      for (int ni = 0; ni < 2; ++ni)
        acc[mi][ni] = wmma_bf16(af[mi], bfr[ni], acc[mi][ni]);
  }

  const int half = lane >> 4;
  const int ncol = lane & 15;
#pragma unroll
  for (int mi = 0; mi < 4; ++mi)
#pragma unroll
    for (int ni = 0; ni < 2; ++ni)
#pragma unroll
      for (int r = 0; r < 8; ++r) {
        const int row = m0 + wm + mi * 16 + half * 8 + r;
        const int col = n0 + wn + ni * 16 + ncol;
        out[(size_t)row * Cch + col] = acc[mi][ni][r] + bp[col];
      }
}

// ---------------------------------------------------------------------------
extern "C" void kernel_launch(void* const* d_in, const int* in_sizes, int n_in,
                              void* d_out, int out_size, void* d_ws, size_t ws_size,
                              hipStream_t stream) {
  const float* x      = (const float*)d_in[0];
  // d_in[1] = xpos (unused: rope disabled in this configuration)
  const float* W_qkv  = (const float*)d_in[2];
  const float* b_qkv  = (const float*)d_in[3];
  const float* W_proj = (const float*)d_in[4];
  const float* b_proj = (const float*)d_in[5];
  float*       out    = (float*)d_out;

  const size_t per = (size_t)Bsz * Hh * Nsq * HD;  // 3,145,728 bf16 elems
  bf16* qws = (bf16*)d_ws;
  bf16* kws = qws + per;
  bf16* vws = kws + per;
  bf16* aws = vws + per;  // attn result [B,N,C]  (total ws use: 24 MiB)

  // 1) QKV projection (2304 x 4096 output, 128x128 tiles)
  qkv_gemm_kernel<<<dim3((3 * Cch) / 128, (Bsz * Nsq) / 128), 256, 0, stream>>>(
      x, W_qkv, b_qkv, qws, kws, vws);

  // 2) Flash attention: 16 q-tiles of 128 rows x 24 (b,h) pairs
  flash_attn_kernel<<<dim3(Nsq / 128, Bsz * Hh), 256, 0, stream>>>(
      qws, kws, vws, aws);

  // 3) Output projection (768 x 4096 output)
  proj_gemm_kernel<<<dim3(Cch / 128, (Bsz * Nsq) / 128), 256, 0, stream>>>(
      aws, W_proj, b_proj, out);
}
